// MultiHeadAttention_23828478558257
// MI455X (gfx1250) — matrix operand order repaired and checked by reading, not compiled, not run on hardware
//
#include <hip/hip_runtime.h>
#include <hip/hip_bf16.h>
#include <stdint.h>

typedef __bf16 bf16_t;
typedef __attribute__((ext_vector_type(16))) __bf16 v16bf;
typedef __attribute__((ext_vector_type(8)))  __bf16 v8bf;
typedef __attribute__((ext_vector_type(4)))  __bf16 v4bf;
typedef __attribute__((ext_vector_type(8)))  float   v8f;
typedef __attribute__((ext_vector_type(4)))  float   v4f;

#define BATCH 2
#define SEQ   2048
#define DIM   1024
#define HEADS 16
#define DK    64
#define MTOK  (BATCH * SEQ)   // 4096 tokens

// ---------------------------------------------------------------------------
// CDNA5 async global->LDS staging (ASYNCcnt path, CDNA5 ISA)
// ---------------------------------------------------------------------------
__device__ __forceinline__ void async_ld_b128(void* lds_dst, const void* src) {
  // Low 32 bits of a generic shared pointer = workgroup-relative LDS offset
  // (flat LDS aperture, ISA 10.2).
  uint32_t lds_off = (uint32_t)(uintptr_t)lds_dst;
  asm volatile("global_load_async_to_lds_b128 %0, %1, off"
               :: "v"(lds_off), "v"(src)
               : "memory");
}
__device__ __forceinline__ void wait_async_le1() {
  asm volatile("s_wait_asynccnt 0x1" ::: "memory");
}
__device__ __forceinline__ void wait_async_le2() {
  asm volatile("s_wait_asynccnt 0x2" ::: "memory");
}

// ---------------------------------------------------------------------------
// WMMA helpers (gfx1250, wave32). Layouts per CDNA5 ISA §7.12.2.
// ---------------------------------------------------------------------------
__device__ __forceinline__ v8f wmma_bf16(v16bf a, v16bf b, v8f c) {
  // (neg_a, A, neg_b, B, c_mod, C, reuse_a, reuse_b)
  return __builtin_amdgcn_wmma_f32_16x16x32_bf16(
      false, a, false, b, (short)0, c, false, false);
}

// A-matrix fragment: 16x32 bf16, src row-major, row stride `ld` elements.
// Lane L: row M = L%16, half h = L/16; elems 0..7 -> K=h*8+j, 8..15 -> K=16+h*8+j.
__device__ __forceinline__ v16bf load_frag_a(const bf16_t* base, int ld) {
  int lane = threadIdx.x & 31;
  int m = lane & 15;
  int h = lane >> 4;
  const bf16_t* p = base + (size_t)m * ld + h * 8;
  v8bf lo = *reinterpret_cast<const v8bf*>(p);
  v8bf hi = *reinterpret_cast<const v8bf*>(p + 16);
  v16bf r;
#pragma unroll
  for (int i = 0; i < 8; ++i) { r[i] = lo[i]; r[i + 8] = hi[i]; }
  return r;
}

// B-matrix fragment: 32(K) x 16(N) bf16 from [N][K] row-major storage,
// row stride `ld`. Lane L: N = L%16, h = L/16; elems 0..7 -> K=h*8+j, 8..15 -> K=16+h*8+j.
__device__ __forceinline__ v16bf load_frag_b(const bf16_t* base, int ld) {
  int lane = threadIdx.x & 31;
  int n = lane & 15;
  int h = lane >> 4;
  const bf16_t* p = base + (size_t)n * ld + h * 8;
  v8bf lo = *reinterpret_cast<const v8bf*>(p);
  v8bf hi = *reinterpret_cast<const v8bf*>(p + 16);
  v16bf r;
#pragma unroll
  for (int i = 0; i < 8; ++i) { r[i] = lo[i]; r[i + 8] = hi[i]; }
  return r;
}

// ---------------------------------------------------------------------------
// Kernel 1: f32 -> bf16 elementwise convert (x4 vectorized)
// ---------------------------------------------------------------------------
__global__ void mha_cvt_bf16(const float* __restrict__ in,
                             bf16_t* __restrict__ out, int n4) {
  int i = blockIdx.x * blockDim.x + threadIdx.x;
  if (i >= n4) return;
  v4f x = *reinterpret_cast<const v4f*>(in + (size_t)i * 4);
  v4bf y;
#pragma unroll
  for (int j = 0; j < 4; ++j) y[j] = (bf16_t)x[j];
  *reinterpret_cast<v4bf*>(out + (size_t)i * 4) = y;
}

// ---------------------------------------------------------------------------
// Kernel 2: transpose + convert weights: Wt[n][k] = (bf16)W[k][n], D x D
// ---------------------------------------------------------------------------
__global__ void mha_transpose_bf16(const float* __restrict__ W,
                                   bf16_t* __restrict__ Wt) {
  __shared__ float tile[16][17];
  int k = blockIdx.y * 16 + threadIdx.y;
  int n = blockIdx.x * 16 + threadIdx.x;
  tile[threadIdx.y][threadIdx.x] = W[(size_t)k * DIM + n];
  __syncthreads();
  int nn = blockIdx.x * 16 + threadIdx.y;
  int kk = blockIdx.y * 16 + threadIdx.x;
  Wt[(size_t)nn * DIM + kk] = (bf16_t)tile[threadIdx.x][threadIdx.y];
}

// ---------------------------------------------------------------------------
// Kernel 3/5: GEMM  C[4096][1024] = A[M][K] * Wt[N][K]^T + bias
// Block: 256 threads (8 waves) -> 256 rows x 64 cols. Shared Wt K-slab
// (64x32 bf16 = 4KB) double-buffered in LDS via async global->LDS loads.
// Each wave: 2 A-frags (global b128), 4 B-frags (LDS b128), 8 WMMAs / K-step.
//   mode 0: bf16 row-major (Q,K)   mode 1: f32 row-major (final)
//   mode 2: bf16 per-head transposed Vt[b][h][d][s] (V)
// ---------------------------------------------------------------------------
__global__ void mha_gemm_bf16(const bf16_t* __restrict__ A,
                              const bf16_t* __restrict__ Bt,
                              const float* __restrict__ bias,
                              bf16_t* __restrict__ outb,
                              float* __restrict__ outf,
                              int mode) {
  __shared__ bf16_t Bs[2][64 * 32];  // [buf][n][k], row = 32 elems (64B)

  const int tid = threadIdx.x;
  const int wave = tid >> 5;
  const int lane = tid & 31;
  const int n16 = lane & 15;
  const int half = lane >> 4;
  const int row0 = blockIdx.y * 256 + wave * 32;
  const int col0 = blockIdx.x * 64;

  // staging decomposition: 256 threads x 16B cover the 4KB tile
  const int bn = tid >> 2;         // 0..63 : tile row (N)
  const int bc = (tid & 3) * 8;    // chunk offset in elements

  // prologue: stage K-slab 0 into buffer 0
  async_ld_b128(&Bs[0][bn * 32 + bc], Bt + (size_t)(col0 + bn) * DIM + bc);

  v8f acc[8] = {};
  for (int kk = 0; kk < DIM; kk += 32) {
    const int buf = (kk >> 5) & 1;
    const int knext = (kk + 32) & (DIM - 1);  // wrap: harmless extra prefetch
    async_ld_b128(&Bs[buf ^ 1][bn * 32 + bc],
                  Bt + (size_t)(col0 + bn) * DIM + knext + bc);
    wait_async_le1();          // current tile's async op (this wave) complete
    __syncthreads();           // all waves' staging visible

    v16bf a0 = load_frag_a(A + (size_t)row0 * DIM + kk, DIM);
    v16bf a1 = load_frag_a(A + (size_t)(row0 + 16) * DIM + kk, DIM);
#pragma unroll
    for (int t = 0; t < 4; ++t) {
      v16bf b = load_frag_b(&Bs[buf][(16 * t) * 32], 32);
      acc[t]     = wmma_bf16(a0, b, acc[t]);
      acc[t + 4] = wmma_bf16(a1, b, acc[t + 4]);
    }
    __syncthreads();           // protect buf^1 before next iteration's issue
  }

#pragma unroll
  for (int g = 0; g < 2; ++g) {
#pragma unroll
    for (int t = 0; t < 4; ++t) {
      int col = col0 + 16 * t + n16;
      float bv = bias[col];
#pragma unroll
      for (int r = 0; r < 8; ++r) {
        int row = row0 + 16 * g + r + 8 * half;
        float val = acc[t + 4 * g][r] + bv;
        if (mode == 1) {
          outf[(size_t)row * DIM + col] = val;
        } else if (mode == 0) {
          outb[(size_t)row * DIM + col] = (bf16_t)val;
        } else {  // mode 2: Vt[b][head][d][s]
          int bb = row >> 11, ss = row & (SEQ - 1);
          int head = col >> 6, dd = col & (DK - 1);
          outb[(((size_t)bb * HEADS + head) * DK + dd) * SEQ + ss] =
              (bf16_t)val;
        }
      }
    }
  }
}

// ---------------------------------------------------------------------------
// Kernel 4: flash attention per (b, head).  Qb/Kb: [B*S][D] bf16, Vt:
// [B][H][DK][S] bf16.  Block = 8 waves x 16 query rows = 128 queries.
// Shared K-tile (32x64) and V-tile (64x32) double-buffered via async
// global->LDS; all 8 waves consume the same tiles (8x traffic reduction).
// Online softmax with __shfl_xor row reductions; P re-fragmented through a
// wave-private LDS slab.  Quirk preserved: scale = 1/sqrt(S).
// ---------------------------------------------------------------------------
__global__ void mha_flash_attn(const bf16_t* __restrict__ Qb,
                               const bf16_t* __restrict__ Kb,
                               const bf16_t* __restrict__ Vt,
                               bf16_t* __restrict__ Ob) {
  __shared__ bf16_t Ks[2][32 * 64];  // [buf][kpos_rel][d], row = 64 elems
  __shared__ bf16_t Vs[2][64 * 32];  // [buf][d][kpos_rel], row = 32 elems
  __shared__ bf16_t Ps[8][16 * 32];  // wave-private P slabs

  const int tid = threadIdx.x;
  const int wave = tid >> 5;
  const int lane = tid & 31;
  const int n16 = lane & 15;
  const int half = lane >> 4;

  const int bh = blockIdx.y;
  const int b = bh >> 4;
  const int head = bh & 15;
  const int q0 = blockIdx.x * 128 + wave * 16;

  const bf16_t* Qh = Qb + (size_t)b * SEQ * DIM + head * DK;
  const bf16_t* Kh = Kb + (size_t)b * SEQ * DIM + head * DK;
  const bf16_t* Vh = Vt + ((size_t)b * HEADS + head) * DK * SEQ;
  bf16_t* Oh = Ob + (size_t)b * SEQ * DIM + head * DK;
  bf16_t* myP = &Ps[wave][0];

  const float scale = 0.022097086912079612f;  // 1/sqrt(2048)

  // staging decomposition (256 threads x 16B per tile)
  const int kn = tid >> 3, kc = (tid & 7) * 8;  // K tile: 32 rows x 64 cols
  const int vn = tid >> 2, vc = (tid & 3) * 8;  // V tile: 64 rows x 32 cols

  // prologue: stage tile 0
  async_ld_b128(&Ks[0][kn * 64 + kc], Kh + (size_t)kn * DIM + kc);
  async_ld_b128(&Vs[0][vn * 32 + vc], Vh + (size_t)vn * SEQ + vc);

  // Q fragments (d in [0,32) and [32,64)) resident in registers
  v16bf qa0 = load_frag_a(Qh + (size_t)q0 * DIM + 0, DIM);
  v16bf qa1 = load_frag_a(Qh + (size_t)q0 * DIM + 32, DIM);

  v8f oacc[4] = {};
  float m[8], l[8];
#pragma unroll
  for (int r = 0; r < 8; ++r) { m[r] = -INFINITY; l[r] = 0.0f; }

  for (int k0 = 0; k0 < SEQ; k0 += 32) {
    const int buf = (k0 >> 5) & 1;
    const int knext = (k0 + 32) & (SEQ - 1);  // wrap prefetch, race-free
    async_ld_b128(&Ks[buf ^ 1][kn * 64 + kc],
                  Kh + (size_t)(knext + kn) * DIM + kc);
    async_ld_b128(&Vs[buf ^ 1][vn * 32 + vc],
                  Vh + (size_t)vn * SEQ + knext + vc);
    wait_async_le2();          // current tile's 2 async ops complete
    __syncthreads();

    // scores: two 16x16 tiles (key cols k0..+16, k0+16..+32), K-dim = 64
    v8f s0 = {}, s1 = {};
    {
      v16bf kb;
      kb = load_frag_b(&Ks[buf][0 * 64 + 0], 64);
      s0 = wmma_bf16(qa0, kb, s0);
      kb = load_frag_b(&Ks[buf][0 * 64 + 32], 64);
      s0 = wmma_bf16(qa1, kb, s0);
      kb = load_frag_b(&Ks[buf][16 * 64 + 0], 64);
      s1 = wmma_bf16(qa0, kb, s1);
      kb = load_frag_b(&Ks[buf][16 * 64 + 32], 64);
      s1 = wmma_bf16(qa1, kb, s1);
    }

    // online softmax (row r+8*half lives on this 16-lane group)
    float corr[8];
#pragma unroll
    for (int r = 0; r < 8; ++r) {
      float a0 = s0[r] * scale;
      float a1 = s1[r] * scale;
      float mx = fmaxf(a0, a1);
#pragma unroll
      for (int off = 1; off < 16; off <<= 1)
        mx = fmaxf(mx, __shfl_xor(mx, off, 16));
      float mnew = fmaxf(m[r], mx);
      float c = __expf(m[r] - mnew);
      float p0 = __expf(a0 - mnew);
      float p1 = __expf(a1 - mnew);
      float rs = p0 + p1;
#pragma unroll
      for (int off = 1; off < 16; off <<= 1)
        rs += __shfl_xor(rs, off, 16);
      l[r] = l[r] * c + rs;
      m[r] = mnew;
      corr[r] = c;
      myP[(r + 8 * half) * 32 + n16] = (bf16_t)p0;
      myP[(r + 8 * half) * 32 + 16 + n16] = (bf16_t)p1;
    }

#pragma unroll
    for (int t = 0; t < 4; ++t)
#pragma unroll
      for (int r = 0; r < 8; ++r) oacc[t][r] *= corr[r];

    // P @ V from LDS
    v16bf pa = load_frag_a(myP, 32);
#pragma unroll
    for (int t = 0; t < 4; ++t) {
      v16bf vb = load_frag_b(&Vs[buf][(16 * t) * 32], 32);
      oacc[t] = wmma_bf16(pa, vb, oacc[t]);
    }
    __syncthreads();           // protect buf^1 before next iteration's issue
  }

  // normalize + store merged-head attn (bf16) for the output projection
#pragma unroll
  for (int r = 0; r < 8; ++r) {
    float inv = 1.0f / l[r];
    int row = q0 + r + 8 * half;
#pragma unroll
    for (int t = 0; t < 4; ++t) {
      Oh[(size_t)row * DIM + 16 * t + n16] = (bf16_t)(oacc[t][r] * inv);
    }
  }
}

// ---------------------------------------------------------------------------
// Host launcher
// ---------------------------------------------------------------------------
extern "C" void kernel_launch(void* const* d_in, const int* in_sizes, int n_in,
                              void* d_out, int out_size, void* d_ws,
                              size_t ws_size, hipStream_t stream) {
  (void)in_sizes; (void)n_in; (void)out_size; (void)ws_size;

  const float* Xq_f = (const float*)d_in[0];
  const float* Xk_f = (const float*)d_in[1];
  const float* Xv_f = (const float*)d_in[2];
  const float* Wq = (const float*)d_in[3];
  const float* bq = (const float*)d_in[4];
  const float* Wk = (const float*)d_in[5];
  const float* bk = (const float*)d_in[6];
  const float* Wv = (const float*)d_in[7];
  const float* bv = (const float*)d_in[8];
  const float* Wo = (const float*)d_in[9];
  const float* bo = (const float*)d_in[10];
  float* out = (float*)d_out;

  // Workspace carve-up (bf16 buffers), total 64 MB.
  char* ws = (char*)d_ws;
  const size_t XBYTES = (size_t)MTOK * DIM * 2;  // 8 MB
  const size_t WBYTES = (size_t)DIM * DIM * 2;   // 2 MB
  bf16_t* Xq = (bf16_t*)ws; ws += XBYTES;
  bf16_t* Xk = (bf16_t*)ws; ws += XBYTES;
  bf16_t* Xv = (bf16_t*)ws; ws += XBYTES;
  bf16_t* Wqt = (bf16_t*)ws; ws += WBYTES;
  bf16_t* Wkt = (bf16_t*)ws; ws += WBYTES;
  bf16_t* Wvt = (bf16_t*)ws; ws += WBYTES;
  bf16_t* Wot = (bf16_t*)ws; ws += WBYTES;
  bf16_t* Qb = (bf16_t*)ws; ws += XBYTES;
  bf16_t* Kb = (bf16_t*)ws; ws += XBYTES;
  bf16_t* Vt = (bf16_t*)ws; ws += XBYTES;   // [B][H][DK][S]
  bf16_t* At = (bf16_t*)ws; ws += XBYTES;   // merged-head attention (bf16)

  // 1) activation converts
  {
    int n4 = (MTOK * DIM) / 4;
    dim3 g((n4 + 255) / 256), blk(256);
    mha_cvt_bf16<<<g, blk, 0, stream>>>(Xq_f, Xq, n4);
    mha_cvt_bf16<<<g, blk, 0, stream>>>(Xk_f, Xk, n4);
    mha_cvt_bf16<<<g, blk, 0, stream>>>(Xv_f, Xv, n4);
  }

  // 2) weight transpose-converts: Wt[n][k] = W[k][n]
  {
    dim3 g(DIM / 16, DIM / 16), blk(16, 16);
    mha_transpose_bf16<<<g, blk, 0, stream>>>(Wq, Wqt);
    mha_transpose_bf16<<<g, blk, 0, stream>>>(Wk, Wkt);
    mha_transpose_bf16<<<g, blk, 0, stream>>>(Wv, Wvt);
    mha_transpose_bf16<<<g, blk, 0, stream>>>(Wo, Wot);
  }

  // 3) QKV projections (V stored per-head transposed)
  {
    dim3 g(DIM / 64, MTOK / 256), blk(256);
    mha_gemm_bf16<<<g, blk, 0, stream>>>(Xq, Wqt, bq, Qb, nullptr, 0);
    mha_gemm_bf16<<<g, blk, 0, stream>>>(Xk, Wkt, bk, Kb, nullptr, 0);
    mha_gemm_bf16<<<g, blk, 0, stream>>>(Xv, Wvt, bv, Vt, nullptr, 2);
  }

  // 4) attention
  {
    dim3 g(SEQ / 128, BATCH * HEADS), blk(256);
    mha_flash_attn<<<g, blk, 0, stream>>>(Qb, Kb, Vt, At);
  }

  // 5) output projection -> f32 out
  {
    dim3 g(DIM / 64, MTOK / 256), blk(256);
    mha_gemm_bf16<<<g, blk, 0, stream>>>(At, Wot, bo, nullptr, out, 1);
  }
}
